// BigramLanguageModel_12841952215258
// MI455X (gfx1250) — matrix operand order repaired and checked by reading, not compiled, not run on hardware
//
#include <hip/hip_runtime.h>

typedef __bf16 bf16;
typedef __attribute__((ext_vector_type(16))) __bf16 v16bf;
typedef __attribute__((ext_vector_type(8)))  float  v8f;
typedef __attribute__((ext_vector_type(4)))  unsigned int v4u;
typedef __attribute__((ext_vector_type(2)))  unsigned int v2u;
typedef __attribute__((ext_vector_type(4)))  float v4f;

constexpr int Bc = 2, Tc = 1024, Vc = 32000, Cc = 1024, Hc = 16, Lc = 6, HSc = 64;
constexpr int BT = Bc * Tc;

union FragB16 { v4u u4[2]; unsigned int u[8]; v16bf v; };
union Acc     { v8f v; float f[8]; };

__device__ __forceinline__ v8f wmma_bf16(v16bf a, v16bf b, v8f c) {
  return __builtin_amdgcn_wmma_f32_16x16x32_bf16(false, a, false, b, (short)0, c,
                                                 false, false);
}

// Async global->LDS b128 copy (CDNA5, tracked by ASYNCcnt). INST_OFFSET is
// added to BOTH the LDS dest and the global source address (ISA 15.18), so a
// row copy is one base pair + immediate offsets.
__device__ __forceinline__ void async_ld_b128(unsigned lds_addr, const void* gptr) {
  asm volatile("global_load_async_to_lds_b128 %0, %1, off"
               :: "v"(lds_addr), "v"(gptr));
}
template<int OFF>
__device__ __forceinline__ void async_ld_b128_off(unsigned lds_addr, const void* gptr) {
  asm volatile("global_load_async_to_lds_b128 %0, %1, off offset:%2"
               :: "v"(lds_addr), "v"(gptr), "i"(OFF));
}
__device__ __forceinline__ void wait_async0() {
  asm volatile("s_wait_asynccnt 0" ::: "memory");
}
__device__ __forceinline__ void wait_ds0() {
  asm volatile("s_wait_dscnt 0" ::: "memory");
}
__device__ __forceinline__ unsigned lds_off(const void* p) {
  return (unsigned)(unsigned long long)p;   // low 32 bits of generic = LDS offset
}

// Compile-time XOR lane exchange via ds_swizzle (group-of-32: imm = xor<<10 | 0x1f)
template<int XOR>
__device__ __forceinline__ float swz_xor(float x) {
  return __int_as_float(
      __builtin_amdgcn_ds_swizzle(__float_as_int(x), (XOR << 10) | 0x1f));
}
// max-reduce / sum-reduce across the 16-lane half-wave
__device__ __forceinline__ float redmax16(float x) {
  x = fmaxf(x, swz_xor<1>(x)); x = fmaxf(x, swz_xor<2>(x));
  x = fmaxf(x, swz_xor<4>(x)); x = fmaxf(x, swz_xor<8>(x));
  return x;
}
__device__ __forceinline__ float redsum16(float x) {
  x += swz_xor<1>(x); x += swz_xor<2>(x); x += swz_xor<4>(x); x += swz_xor<8>(x);
  return x;
}
__device__ __forceinline__ float redsum32(float x) {
  x += swz_xor<1>(x); x += swz_xor<2>(x); x += swz_xor<4>(x);
  x += swz_xor<8>(x); x += swz_xor<16>(x);
  return x;
}

// ---------------------------------------------------------------------------
// Generic bf16 GEMM: D[M,N] = op(A[M,K] @ B[K,N] + bias [+ res]),
// A,B bf16 row-major; bias/res fp32; out fp32 or bf16.
// Block 256 threads = 8 waves; block tile 128x128; wave tile 64x32; K-step 32.
// A tile staged with async global->LDS DMA; B tile transposed manually;
// next-iteration tiles prefetched into GL2/WGP$ with global_prefetch.
// ---------------------------------------------------------------------------
template<bool BIAS, bool RELU, bool RES, bool OUTBF>
__global__ __launch_bounds__(256) void gemm_bf16_kernel(
    const bf16* __restrict__ A, const bf16* __restrict__ Bm,
    const float* __restrict__ bias, const float* __restrict__ res,
    void* __restrict__ Dout, int M, int N, int K)
{
  __shared__ bf16 As[128][40];   // (m, k) row-major, +8 pad
  __shared__ bf16 Bs[128][40];   // (n, k) transposed, +8 pad
  const int tid  = threadIdx.x;
  const int wv   = tid >> 5;
  const int lane = tid & 31;
  const int hf   = lane >> 4;
  const int l16  = lane & 15;
  const int wm   = wv >> 2;        // 0..1  (64-row slab)
  const int wn   = wv & 3;         // 0..3  (32-col slab)
  const int bm   = blockIdx.y * 128;
  const int bn   = blockIdx.x * 128;

  Acc acc[4][2];
  #pragma unroll
  for (int i = 0; i < 4; ++i)
    #pragma unroll
    for (int j = 0; j < 2; ++j)
      #pragma unroll
      for (int r = 0; r < 8; ++r) acc[i][j].f[r] = 0.f;

  const int arow = tid >> 2;            // A stage: 4 threads per row
  const int akc  = (tid & 3) * 8;
  const int bkp  = tid >> 5;            // B stage: k pair
  const int bnq  = (tid & 31) * 4;      // B stage: n quad

  for (int k0 = 0; k0 < K; k0 += 32) {
    // ---- stage A tile 128x32 via async b128 DMA ----
    #pragma unroll
    for (int it = 0; it < 2; ++it) {
      int row = it * 64 + arow;
      async_ld_b128(lds_off(&As[row][akc]),
                    A + (size_t)(bm + row) * K + k0 + akc);
    }
    // ---- stage B tile 32x128, transposed into Bs[n][k] ----
    #pragma unroll
    for (int it = 0; it < 2; ++it) {
      int kp = it * 8 + bkp;
      const bf16* srcp = Bm + (size_t)(k0 + 2 * kp) * N + bn + bnq;
      union { v2u w; bf16 h[4]; } r0, r1;
      r0.w = *(const v2u*)srcp;
      r1.w = *(const v2u*)(srcp + N);
      #pragma unroll
      for (int q4 = 0; q4 < 4; ++q4) {
        Bs[bnq + q4][2 * kp]     = r0.h[q4];
        Bs[bnq + q4][2 * kp + 1] = r1.h[q4];
      }
    }
    // ---- prefetch next k-step tiles while this one is consumed ----
    if (k0 + 32 < K) {
      __builtin_prefetch(A + (size_t)(bm + arow) * K + (k0 + 32) + akc, 0, 1);
      __builtin_prefetch(Bm + (size_t)(k0 + 32 + 2 * bkp) * N + bn + bnq, 0, 1);
    }
    wait_async0();
    __syncthreads();

    // ---- prefetch all fragments, then burst the WMMAs ----
    FragB16 bfrag[2], afrag[4];
    #pragma unroll
    for (int j = 0; j < 2; ++j) {
      const v4u* p = (const v4u*)(&Bs[wn * 32 + j * 16 + l16][0]);
      bfrag[j].u4[0] = p[hf];           // k = hf*8 + 0..7  (u32 pairs)
      bfrag[j].u4[1] = p[2 + hf];       // k = 16 + hf*8 + 0..7
    }
    #pragma unroll
    for (int i = 0; i < 4; ++i) {
      const v4u* p = (const v4u*)(&As[wm * 64 + i * 16 + l16][0]);
      afrag[i].u4[0] = p[hf];           // k pairs: hf*8 + {0,2,4,6}
      afrag[i].u4[1] = p[2 + hf];       // k pairs: 16 + hf*8 + {0,2,4,6}
    }
    #pragma unroll
    for (int i = 0; i < 4; ++i)
      #pragma unroll
      for (int j = 0; j < 2; ++j)
        acc[i][j].v = wmma_bf16(afrag[i].v, bfrag[j].v, acc[i][j].v);
    __syncthreads();
  }

  // ---- epilogue ----
  #pragma unroll
  for (int i = 0; i < 4; ++i)
    #pragma unroll
    for (int j = 0; j < 2; ++j)
      #pragma unroll
      for (int r = 0; r < 8; ++r) {
        int m = bm + wm * 64 + i * 16 + r + 8 * hf;
        int n = bn + wn * 32 + j * 16 + l16;
        float val = acc[i][j].f[r];
        if (BIAS) val += bias[n];
        if (RELU) val = fmaxf(val, 0.f);
        if (RES)  val += res[(size_t)m * N + n];
        if (OUTBF) ((bf16*)Dout)[(size_t)m * N + n] = (bf16)val;
        else       ((float*)Dout)[(size_t)m * N + n] = val;
      }
}

// ---------------------------------------------------------------------------
// Fused causal flash attention. q,k,v,o: bf16 (B,T,H*HS) with head offset.
// Block = 128 threads = 4 independent waves; each wave owns 16 query rows.
// K/V tiles staged with async global->LDS DMA using immediate offsets.
// ---------------------------------------------------------------------------
__global__ __launch_bounds__(128) void attn_kernel(
    const bf16* __restrict__ q, const bf16* __restrict__ k,
    const bf16* __restrict__ v, bf16* __restrict__ o)
{
  __shared__ bf16 Kt[4][32][72];   // per-wave K tile (key-major)
  __shared__ bf16 Vt[4][32][72];   // per-wave V tile (key-major)
  __shared__ bf16 Pt[4][16][40];   // per-wave P tile (row-major 16x32)
  const int wv   = threadIdx.x >> 5;
  const int lane = threadIdx.x & 31;
  const int hf   = lane >> 4;
  const int l16  = lane & 15;
  const int b    = blockIdx.x / Hc;
  const int hd   = blockIdx.x % Hc;
  const int qt   = blockIdx.y * 64 + wv * 16;
  const size_t base = (size_t)b * Tc * Cc + hd * HSc;

  // Q fragments (A-matrix 16x32, two K-slices of the 64-wide head dim)
  FragB16 qf[2];
  {
    const v4u* p = (const v4u*)(q + base + (size_t)(qt + l16) * Cc);
    #pragma unroll
    for (int f = 0; f < 2; ++f) {
      qf[f].u4[0] = p[f * 4 + hf];
      qf[f].u4[1] = p[f * 4 + 2 + hf];
    }
  }

  Acc O[4];
  float mrow[8], lrow[8];
  #pragma unroll
  for (int r = 0; r < 8; ++r) { mrow[r] = -3.0e38f; lrow[r] = 0.f; }
  #pragma unroll
  for (int f = 0; f < 4; ++f)
    #pragma unroll
    for (int r = 0; r < 8; ++r) O[f].f[r] = 0.f;

  const unsigned kl = lds_off(&Kt[wv][lane][0]);
  const unsigned vl = lds_off(&Vt[wv][lane][0]);

  const int ntiles = (qt + 16 + 31) >> 5;
  for (int ktile = 0; ktile < ntiles; ++ktile) {
    const int s0 = ktile * 32;
    // stage K/V tile asynchronously: lane <-> key row, 128B per row,
    // one base address + immediate offsets (offset applies to LDS + global)
    {
      const char* kg = (const char*)(k + base + (size_t)(s0 + lane) * Cc);
      const char* vg = (const char*)(v + base + (size_t)(s0 + lane) * Cc);
      async_ld_b128_off<  0>(kl, kg);  async_ld_b128_off<  0>(vl, vg);
      async_ld_b128_off< 16>(kl, kg);  async_ld_b128_off< 16>(vl, vg);
      async_ld_b128_off< 32>(kl, kg);  async_ld_b128_off< 32>(vl, vg);
      async_ld_b128_off< 48>(kl, kg);  async_ld_b128_off< 48>(vl, vg);
      async_ld_b128_off< 64>(kl, kg);  async_ld_b128_off< 64>(vl, vg);
      async_ld_b128_off< 80>(kl, kg);  async_ld_b128_off< 80>(vl, vg);
      async_ld_b128_off< 96>(kl, kg);  async_ld_b128_off< 96>(vl, vg);
      async_ld_b128_off<112>(kl, kg);  async_ld_b128_off<112>(vl, vg);
    }
    wait_async0();

    // S = Q @ K^T  (two 16x16 C-fragments covering 32 keys)
    FragB16 kf[2][2];
    #pragma unroll
    for (int c = 0; c < 2; ++c) {
      const v4u* p = (const v4u*)(&Kt[wv][c * 16 + l16][0]);
      #pragma unroll
      for (int f = 0; f < 2; ++f) {
        kf[c][f].u4[0] = p[f * 4 + hf];
        kf[c][f].u4[1] = p[f * 4 + 2 + hf];
      }
    }
    Acc S[2];
    #pragma unroll
    for (int c = 0; c < 2; ++c) {
      #pragma unroll
      for (int r = 0; r < 8; ++r) S[c].f[r] = 0.f;
      #pragma unroll
      for (int f = 0; f < 2; ++f)
        S[c].v = wmma_bf16(qf[f].v, kf[c][f].v, S[c].v);
    }

    // online softmax (row r+8*hf lives in VGPR r; columns across 16 lanes)
    #pragma unroll
    for (int r = 0; r < 8; ++r) {
      const int qrow = qt + r + 8 * hf;
      float a0 = S[0].f[r] * 0.125f;
      float a1 = S[1].f[r] * 0.125f;
      if (s0 + l16      > qrow) a0 = -3.0e38f;
      if (s0 + 16 + l16 > qrow) a1 = -3.0e38f;
      const float mx    = redmax16(fmaxf(a0, a1));
      const float mnew  = fmaxf(mrow[r], mx);
      const float alpha = __expf(mrow[r] - mnew);
      const float p0 = __expf(a0 - mnew);
      const float p1 = __expf(a1 - mnew);
      const float ps = redsum16(p0 + p1);
      lrow[r] = lrow[r] * alpha + ps;
      mrow[r] = mnew;
      #pragma unroll
      for (int f = 0; f < 4; ++f) O[f].f[r] *= alpha;
      S[0].f[r] = p0; S[1].f[r] = p1;
    }

    // re-lay P through LDS into A-fragment form (bf16)
    #pragma unroll
    for (int c = 0; c < 2; ++c)
      #pragma unroll
      for (int r = 0; r < 8; ++r)
        Pt[wv][r + 8 * hf][c * 16 + l16] = (bf16)S[c].f[r];
    wait_ds0();
    FragB16 pf;
    {
      const v4u* p = (const v4u*)(&Pt[wv][l16][0]);
      pf.u4[0] = p[hf];
      pf.u4[1] = p[2 + hf];
    }

    // O += P @ V  (B[k=key][n=d], strided gather from key-major Vt)
    FragB16 vf[4];
    #pragma unroll
    for (int f = 0; f < 4; ++f)
      #pragma unroll
      for (int j2 = 0; j2 < 8; ++j2) {
        unsigned lo = *(const unsigned short*)&Vt[wv][hf * 8 + 2 * j2 + (j2 & 4) * 2    ][f * 16 + l16];
        unsigned hi = *(const unsigned short*)&Vt[wv][hf * 8 + 2 * j2 + (j2 & 4) * 2 + 1][f * 16 + l16];
        vf[f].u[j2] = lo | (hi << 16);
      }
    #pragma unroll
    for (int f = 0; f < 4; ++f)
      O[f].v = wmma_bf16(pf.v, vf[f].v, O[f].v);
  }

  // normalize + store
  #pragma unroll
  for (int f = 0; f < 4; ++f)
    #pragma unroll
    for (int r = 0; r < 8; ++r) {
      int t = qt + r + 8 * hf;
      int d = f * 16 + l16;
      o[base + (size_t)t * Cc + d] = (bf16)(O[f].f[r] / lrow[r]);
    }
}

// ---------------------------------------------------------------------------
// LayerNorm: fp32 in -> bf16 out, one block of 128 threads per row (C=1024)
// ---------------------------------------------------------------------------
__global__ __launch_bounds__(128) void ln_kernel(
    const float* __restrict__ x, const float* __restrict__ s,
    const float* __restrict__ b, bf16* __restrict__ out)
{
  const int row = blockIdx.x;
  const float* xr = x + (size_t)row * Cc;
  const int t8 = threadIdx.x * 8;
  v4f v0 = *(const v4f*)(xr + t8);
  v4f v1 = *(const v4f*)(xr + t8 + 4);
  float vals[8] = {v0.x, v0.y, v0.z, v0.w, v1.x, v1.y, v1.z, v1.w};
  float sum = 0.f;
  #pragma unroll
  for (int i = 0; i < 8; ++i) sum += vals[i];
  sum = redsum32(sum);
  __shared__ float red[8];
  const int wv = threadIdx.x >> 5, lane = threadIdx.x & 31;
  if (lane == 0) red[wv] = sum;
  __syncthreads();
  const float mean = (red[0] + red[1] + red[2] + red[3]) * (1.f / Cc);
  float vsum = 0.f;
  #pragma unroll
  for (int i = 0; i < 8; ++i) { float dd = vals[i] - mean; vsum += dd * dd; }
  vsum = redsum32(vsum);
  if (lane == 0) red[4 + wv] = vsum;
  __syncthreads();
  const float inv = rsqrtf((red[4] + red[5] + red[6] + red[7]) * (1.f / Cc) + 1e-5f);
  #pragma unroll
  for (int i = 0; i < 8; ++i) {
    int c = t8 + i;
    out[(size_t)row * Cc + c] = (bf16)((vals[i] - mean) * inv * s[c] + b[c]);
  }
}

// ---------------------------------------------------------------------------
__global__ void embed_kernel(const int* __restrict__ idx,
    const float* __restrict__ tok, const float* __restrict__ pos,
    float* __restrict__ x)
{
  const int bt = blockIdx.x;
  const int t  = bt % Tc;
  const int tk = idx[bt];
  for (int c = threadIdx.x; c < Cc; c += blockDim.x)
    x[(size_t)bt * Cc + c] = tok[(size_t)tk * Cc + c] + pos[(size_t)t * Cc + c];
}

// Wq/Wk/Wv: (L,H,C,HS) fp32 -> per-layer row-major (C, H*HS) bf16
__global__ void pack_qkv_kernel(const float* __restrict__ w,
                                bf16* __restrict__ out, int total)
{
  int i = blockIdx.x * 256 + threadIdx.x;
  if (i >= total) return;
  int d  = i & 63;
  int c  = (i >> 6) & 1023;
  int hd = (i >> 16) & 15;
  int l  = i >> 20;
  out[((size_t)l * Cc + c) * Cc + hd * HSc + d] = (bf16)w[i];
}

__global__ void conv_kernel(const float* __restrict__ w,
                            bf16* __restrict__ out, long long n)
{
  long long i = (long long)blockIdx.x * blockDim.x + threadIdx.x;
  long long stride = (long long)gridDim.x * blockDim.x;
  for (; i < n; i += stride) out[i] = (bf16)w[i];
}

// ---------------------------------------------------------------------------
extern "C" void kernel_launch(void* const* d_in, const int* in_sizes, int n_in,
                              void* d_out, int out_size, void* d_ws, size_t ws_size,
                              hipStream_t stream)
{
  (void)in_sizes; (void)n_in; (void)out_size; (void)ws_size;
  const int*   idx  = (const int*)  d_in[0];
  const float* tok  = (const float*)d_in[1];
  const float* pos  = (const float*)d_in[2];
  const float* Wq   = (const float*)d_in[3];
  const float* Wk   = (const float*)d_in[4];
  const float* Wv   = (const float*)d_in[5];
  const float* Wo   = (const float*)d_in[6];
  const float* bo   = (const float*)d_in[7];
  const float* ln1s = (const float*)d_in[8];
  const float* ln1b = (const float*)d_in[9];
  const float* ln2s = (const float*)d_in[10];
  const float* ln2b = (const float*)d_in[11];
  const float* W1   = (const float*)d_in[12];
  const float* b1   = (const float*)d_in[13];
  const float* W2   = (const float*)d_in[14];
  const float* b2   = (const float*)d_in[15];
  const float* lnfs = (const float*)d_in[16];
  const float* lnfb = (const float*)d_in[17];
  const float* Wlm  = (const float*)d_in[18];
  const float* blm  = (const float*)d_in[19];

  size_t off = 0;
  auto carve = [&](size_t bytes) -> void* {
    void* p = (void*)((char*)d_ws + off);
    off += (bytes + 255) & ~(size_t)255;
    return p;
  };
  float* x   = (float*)carve((size_t)BT * Cc * 4);
  bf16* hb   = (bf16*) carve((size_t)BT * Cc * 2);
  bf16* qb   = (bf16*) carve((size_t)BT * Cc * 2);
  bf16* kb   = (bf16*) carve((size_t)BT * Cc * 2);
  bf16* vb   = (bf16*) carve((size_t)BT * Cc * 2);
  bf16* ab   = (bf16*) carve((size_t)BT * Cc * 2);
  bf16* mid  = (bf16*) carve((size_t)BT * 4 * Cc * 2);
  bf16* WqP  = (bf16*) carve((size_t)Lc * Cc * Cc * 2);
  bf16* WkP  = (bf16*) carve((size_t)Lc * Cc * Cc * 2);
  bf16* WvP  = (bf16*) carve((size_t)Lc * Cc * Cc * 2);
  bf16* WoP  = (bf16*) carve((size_t)Lc * Cc * Cc * 2);
  bf16* W1P  = (bf16*) carve((size_t)Lc * Cc * 4 * Cc * 2);
  bf16* W2P  = (bf16*) carve((size_t)Lc * Cc * 4 * Cc * 2);
  bf16* WlmP = (bf16*) carve((size_t)Cc * Vc * 2);

  const int qkvTotal = Lc * Hc * Cc * HSc;
  pack_qkv_kernel<<<(qkvTotal + 255) / 256, 256, 0, stream>>>(Wq, WqP, qkvTotal);
  pack_qkv_kernel<<<(qkvTotal + 255) / 256, 256, 0, stream>>>(Wk, WkP, qkvTotal);
  pack_qkv_kernel<<<(qkvTotal + 255) / 256, 256, 0, stream>>>(Wv, WvP, qkvTotal);
  conv_kernel<<<4096, 256, 0, stream>>>(Wo,  WoP,  (long long)Lc * Cc * Cc);
  conv_kernel<<<4096, 256, 0, stream>>>(W1,  W1P,  (long long)Lc * Cc * 4 * Cc);
  conv_kernel<<<4096, 256, 0, stream>>>(W2,  W2P,  (long long)Lc * Cc * 4 * Cc);
  conv_kernel<<<4096, 256, 0, stream>>>(Wlm, WlmP, (long long)Cc * Vc);

  embed_kernel<<<BT, 256, 0, stream>>>(idx, tok, pos, x);

  dim3 g1(Cc / 128, BT / 128);
  dim3 g2(4 * Cc / 128, BT / 128);
  for (int l = 0; l < Lc; ++l) {
    ln_kernel<<<BT, 128, 0, stream>>>(x, ln1s + l * Cc, ln1b + l * Cc, hb);
    gemm_bf16_kernel<false,false,false,true><<<g1, 256, 0, stream>>>(
        hb, WqP + (size_t)l * Cc * Cc, nullptr, nullptr, qb, BT, Cc, Cc);
    gemm_bf16_kernel<false,false,false,true><<<g1, 256, 0, stream>>>(
        hb, WkP + (size_t)l * Cc * Cc, nullptr, nullptr, kb, BT, Cc, Cc);
    gemm_bf16_kernel<false,false,false,true><<<g1, 256, 0, stream>>>(
        hb, WvP + (size_t)l * Cc * Cc, nullptr, nullptr, vb, BT, Cc, Cc);
    attn_kernel<<<dim3(Bc * Hc, Tc / 64), 128, 0, stream>>>(qb, kb, vb, ab);
    gemm_bf16_kernel<true,false,true,false><<<g1, 256, 0, stream>>>(
        ab, WoP + (size_t)l * Cc * Cc, bo + l * Cc, x, x, BT, Cc, Cc);
    ln_kernel<<<BT, 128, 0, stream>>>(x, ln2s + l * Cc, ln2b + l * Cc, hb);
    gemm_bf16_kernel<true,true,false,true><<<g2, 256, 0, stream>>>(
        hb, W1P + (size_t)l * Cc * 4 * Cc, b1 + l * 4 * Cc, nullptr, mid,
        BT, 4 * Cc, Cc);
    gemm_bf16_kernel<true,false,true,false><<<g1, 256, 0, stream>>>(
        mid, W2P + (size_t)l * 4 * Cc * Cc, b2 + l * Cc, x, x, BT, Cc, 4 * Cc);
  }
  ln_kernel<<<BT, 128, 0, stream>>>(x, lnfs, lnfb, hb);
  dim3 gl(Vc / 128, BT / 128);
  gemm_bf16_kernel<true,false,false,false><<<gl, 256, 0, stream>>>(
      hb, WlmP, blm, nullptr, (float*)d_out, BT, Vc, Cc);
}
